// RFBMultiHeadAttn_V2_2980707303830
// MI455X (gfx1250) — matrix operand order repaired and checked by reading, not compiled, not run on hardware
//
#include <hip/hip_runtime.h>
#include <hip/hip_bf16.h>

// MI455X / gfx1250, wave32: WMMA bf16 + Tensor Data Mover (TDM) staging.
typedef __attribute__((ext_vector_type(16))) __bf16 v16bf;
typedef __attribute__((ext_vector_type(8)))  __bf16 v8bf;
typedef __attribute__((ext_vector_type(8)))  float  v8f;
typedef __attribute__((ext_vector_type(4)))  unsigned int v4u;
typedef __attribute__((ext_vector_type(8)))  int v8i;
typedef __attribute__((ext_vector_type(4)))  int v4i;

#define NH   5
#define FH   64
#define CIN  64
#define DD   8
#define HWSZ 4096   // H*W
#define CC   4      // feature channels per workgroup
#define NTC  256    // hw columns staged per TDM chunk
#define NCHUNK (HWSZ / NTC)

#if defined(__AMDGCN__) && __has_builtin(__builtin_amdgcn_tensor_load_to_lds)
#define HAVE_TDM 1
#else
#define HAVE_TDM 0
#endif

__device__ __forceinline__ v8f wmma_bf16(v16bf a, v16bf b, v8f c) {
  return __builtin_amdgcn_wmma_f32_16x16x32_bf16(false, a, false, b, (short)0, c,
                                                 false, false);
}

// A operand (16x32 MxK) from row-major bf16, ld=64. Two ds_load_b128.
__device__ __forceinline__ v16bf frag_a_lds(const __bf16* src, int k0, int lane) {
  const int row = lane & 15;
  const int kb  = (lane < 16) ? 0 : 8;
  const __bf16* p = src + row * 64 + k0 + kb;
  v8bf lo = *(const v8bf*)p;
  v8bf hi = *(const v8bf*)(p + 16);
  return __builtin_shufflevector(lo, hi, 0,1,2,3,4,5,6,7,8,9,10,11,12,13,14,15);
}

// B operand (32x16 KxN) with B[k][n] = srcT[n][k], srcT row-major ld=64.
__device__ __forceinline__ v16bf frag_b_lds(const __bf16* srcT, int k0, int lane) {
  const int n  = lane & 15;
  const int kb = (lane < 16) ? 0 : 8;
  const __bf16* p = srcT + n * 64 + k0 + kb;
  v8bf lo = *(const v8bf*)p;
  v8bf hi = *(const v8bf*)(p + 16);
  return __builtin_shufflevector(lo, hi, 0,1,2,3,4,5,6,7,8,9,10,11,12,13,14,15);
}

#if HAVE_TDM
// Issue one 2D TDM transfer: tile [64 cin][NTC hw] f32, row stride 32768 elements.
__device__ __forceinline__ void tdm_issue(const float* gsrc, unsigned lds_byte_off) {
  unsigned long long ga = (unsigned long long)(uintptr_t)gsrc;
  v4u g0;
  g0[0] = 1u;                                          // count=1 (valid), user mode
  g0[1] = lds_byte_off;                                // LDS dest byte address
  g0[2] = (unsigned)(ga & 0xffffffffu);                // global_addr[31:0]
  g0[3] = (unsigned)((ga >> 32) & 0x01ffffffu) | (2u << 30); // addr[56:32] | type=2
  v8i g1;
  g1[0] = (int)(2u << 16);                             // data_size=4B, wg_mask=0
  g1[1] = (int)((HWSZ & 0xffffu) << 16);               // tensor_dim0[15:0]=4096
  g1[2] = (int)((unsigned)CIN << 16);                  // td0 hi=0 | tensor_dim1=64
  g1[3] = (int)((unsigned)NTC << 16);                  // td1 hi=0 | tile_dim0=NTC
  g1[4] = (int)CIN;                                    // tile_dim1=64, tile_dim2=0
  g1[5] = (int)(DD * HWSZ);                            // tensor_dim0_stride=32768
  g1[6] = 0;                                           // stride hi / dim1_stride lo
  g1[7] = 0;
  v4i gz4 = {0, 0, 0, 0};
  v8i gz8 = {0, 0, 0, 0, 0, 0, 0, 0};
  __builtin_amdgcn_tensor_load_to_lds(g0, g1, gz4, gz4, gz8, 0);
}
#endif

__global__ void fused_mha_kernel(const float* __restrict__ x,
                                 const float* __restrict__ Wqkv,
                                 const float* __restrict__ bqkv,
                                 const float* __restrict__ gamma,
                                 float* __restrict__ out) {
  extern __shared__ char smem[];
  __bf16* sQ   = (__bf16*)smem;                       // CC*4096 bf16
  __bf16* sK   = sQ  + CC * HWSZ;
  __bf16* sVT  = sK  + CC * HWSZ;                     // transposed [w][h]
  char*   dynr = smem + (size_t)3 * CC * HWSZ * sizeof(__bf16);
  float*  sX   = (float*)dynr;                        // 2 x (64 x NTC) f32 staging
  __bf16* sAtt = (__bf16*)dynr;                       // overlaps sX (phase 2 only)

  const int lane    = threadIdx.x & 31;
  const int wave    = threadIdx.x >> 5;
  const int halfsel = (lane < 16) ? 0 : 1;
  const int c0 = blockIdx.x * CC;
  const int d  = blockIdx.y / NH;
  const int nh = blockIdx.y % NH;
  const int b  = blockIdx.z;

  // ---------------- Phase 1: QKV projection (W @ x), TDM-staged ----------------
  v16bf aW[2];
  {
    int r = lane & 15;
    int t = r >> 2; if (t > 2) t = 2;                 // rows 12..15 padded
    int o = nh * 192 + t * 64 + (c0 + (r & 3));
    const float* wp = Wqkv + o * CIN;
    const int kb = halfsel ? 16 : 0;
#pragma unroll
    for (int kc = 0; kc < 2; ++kc) {
      v16bf f;
#pragma unroll
      for (int j = 0; j < 8; ++j) {
        f[j]     = (__bf16)wp[kc * 32 + kb + j];
        f[8 + j] = (__bf16)wp[kc * 32 + kb + 8 + j];
      }
      aW[kc] = f;
    }
  }

  float bias8[8];
#pragma unroll
  for (int r = 0; r < 8; ++r) {
    int rr = r + halfsel * 8;
    int t = rr >> 2; if (t > 2) t = 2;
    int o = nh * 192 + t * 64 + (c0 + (rr & 3));
    bias8[r] = bqkv[o];
  }

  const float* xbd = x + (size_t)(b * CIN) * DD * HWSZ + d * HWSZ;

#if HAVE_TDM
  if (wave == 0) tdm_issue(xbd, (unsigned)(uintptr_t)sX);   // prefetch chunk 0
#endif

  for (int ch = 0; ch < NCHUNK; ++ch) {
    float* buf = sX + (size_t)(ch & 1) * CIN * NTC;
#if HAVE_TDM
    if (wave == 0) {
      if (ch + 1 < NCHUNK) {
        float* nbuf = sX + (size_t)((ch + 1) & 1) * CIN * NTC;
        tdm_issue(xbd + (ch + 1) * NTC, (unsigned)(uintptr_t)nbuf);
        __builtin_amdgcn_s_wait_tensorcnt((short)1);
      } else {
        __builtin_amdgcn_s_wait_tensorcnt((short)0);
      }
    }
#else
    // fallback: cooperative global->LDS copy of chunk ch
    for (int idx = threadIdx.x; idx < CIN * NTC; idx += 256) {
      int rr = idx / NTC, cc2 = idx % NTC;
      buf[idx] = xbd[(size_t)rr * DD * HWSZ + ch * NTC + cc2];
    }
#endif
    __syncthreads();   // staged chunk visible to all waves

    // 16 N-tiles per chunk, 2 per wave
#pragma unroll
    for (int tt = 0; tt < 2; ++tt) {
      const int col16 = (wave * 2 + tt) * 16;           // within chunk
      const int col   = ch * NTC + col16 + (lane & 15); // global hw column
      v8f acc = {};
#pragma unroll
      for (int kc = 0; kc < 2; ++kc) {
        const int kb = halfsel ? 16 : 0;
        const float* xp = buf + (kc * 32 + kb) * NTC + col16 + (lane & 15);
        v16bf fb;
#pragma unroll
        for (int j = 0; j < 16; ++j) fb[j] = (__bf16)xp[j * NTC];
        acc = wmma_bf16(aW[kc], fb, acc);
      }
      const int h = col >> 6, w = col & 63;
#pragma unroll
      for (int r = 0; r < 8; ++r) {
        const int rr = r + halfsel * 8;
        const int t = rr >> 2, cidx = rr & 3;
        const __bf16 val = (__bf16)(acc[r] + bias8[r]);
        __bf16* dst = (t == 0) ? sQ : ((t == 1) ? sK : sVT);
        const int rrow = (t == 2) ? w : h;               // V stored transposed
        const int rcol = (t == 2) ? h : w;
        if (rr < 3 * CC) dst[cidx * HWSZ + rrow * 64 + rcol] = val;
      }
    }
    __syncthreads();   // all reads of buf done before it is refilled
  }

  // ---------------- Phase 2: attention per (c, i-tile) ----------------
  for (int pp = 0; pp < 2; ++pp) {
    const int pair = wave + pp * 8;
    const int ci = pair >> 2;
    const int it = pair & 3;
    const __bf16* Kc  = sK  + ci * HWSZ;
    const __bf16* Qc  = sQ  + ci * HWSZ;
    const __bf16* VTc = sVT + ci * HWSZ;
    __bf16* attS = sAtt + wave * 16 * 64;

    v16bf ka[2];
#pragma unroll
    for (int kc = 0; kc < 2; ++kc)
      ka[kc] = frag_a_lds(Kc + it * 16 * 64, kc * 32, lane);

    v8f sim[4];
#pragma unroll
    for (int jt = 0; jt < 4; ++jt) {
      v8f acc = {};
#pragma unroll
      for (int kc = 0; kc < 2; ++kc) {
        v16bf qb = frag_b_lds(Qc + jt * 16 * 64, kc * 32, lane);
        acc = wmma_bf16(ka[kc], qb, acc);
      }
      sim[jt] = acc;
    }

#pragma unroll
    for (int r = 0; r < 8; ++r) {
      float m = -3.402823e38f;
#pragma unroll
      for (int jt = 0; jt < 4; ++jt) m = fmaxf(m, sim[jt][r]);
#pragma unroll
      for (int s = 1; s < 16; s <<= 1) m = fmaxf(m, __shfl_xor(m, s, 32));
      float sum = 0.f;
#pragma unroll
      for (int jt = 0; jt < 4; ++jt) {
        float p = __expf(sim[jt][r] - m);
        sim[jt][r] = p; sum += p;
      }
#pragma unroll
      for (int s = 1; s < 16; s <<= 1) sum += __shfl_xor(sum, s, 32);
      const float inv = 1.f / sum;
#pragma unroll
      for (int jt = 0; jt < 4; ++jt) sim[jt][r] *= inv;
    }

#pragma unroll
    for (int jt = 0; jt < 4; ++jt)
#pragma unroll
      for (int r = 0; r < 8; ++r) {
        const int row = r + halfsel * 8;
        attS[row * 64 + jt * 16 + (lane & 15)] = (__bf16)sim[jt][r];
      }

    v16bf attA[2];
#pragma unroll
    for (int kc = 0; kc < 2; ++kc)
      attA[kc] = frag_a_lds(attS, kc * 32, lane);

    const int c  = c0 + ci;
    const int chn = c * NH + nh;
    const int n2 = chn >> 6, c2 = chn & 63;
    const int oc = c2 * NH + n2;
    const float g = gamma[n2];

#pragma unroll
    for (int wt = 0; wt < 4; ++wt) {
      v8f acc = {};
#pragma unroll
      for (int kc = 0; kc < 2; ++kc) {
        v16bf vb = frag_b_lds(VTc + wt * 16 * 64, kc * 32, lane);
        acc = wmma_bf16(attA[kc], vb, acc);
      }
      const int w = wt * 16 + (lane & 15);
#pragma unroll
      for (int r = 0; r < 8; ++r) {
        const int i = it * 16 + r + halfsel * 8;
        const int oidx = ((b * (CIN * NH) + oc) * DD + d) * HWSZ + i * 64 + w;
        const int xidx = ((b * CIN + c2) * DD + d) * HWSZ + i * 64 + w;
        out[oidx] = g * acc[r] + x[xidx];
      }
    }
  }
}

extern "C" void kernel_launch(void* const* d_in, const int* in_sizes, int n_in,
                              void* d_out, int out_size, void* d_ws, size_t ws_size,
                              hipStream_t stream) {
  const float* x     = (const float*)d_in[0];
  const float* Wqkv  = (const float*)d_in[1];
  const float* bqkv  = (const float*)d_in[2];
  const float* gamma = (const float*)d_in[3];
  float* out = (float*)d_out;

  dim3 grid(FH / CC, DD * NH, 4);
  dim3 block(256);
  size_t shmem = (size_t)3 * CC * HWSZ * sizeof(__bf16)        // Q,K,Vt: 96 KB
               + (size_t)2 * CIN * NTC * sizeof(float);        // x staging: 128 KB
  hipLaunchKernelGGL(fused_mha_kernel, grid, block, shmem, stream,
                     x, Wqkv, bqkv, gamma, out);
  (void)in_sizes; (void)n_in; (void)out_size; (void)d_ws; (void)ws_size;
}
